// GNN_46651934769467
// MI455X (gfx1250) — compile-verified, matrix operand order checked by reading, not run on hardware
//
#include <hip/hip_runtime.h>

#define NN 100000
#define NE 800000

typedef __attribute__((ext_vector_type(16))) __bf16        v16bf;
typedef __attribute__((ext_vector_type(8)))  float         v8f;
typedef __attribute__((ext_vector_type(4)))  unsigned int  v4u;

union Frag { v16bf v; v4u q[2]; };            // one WMMA A/B fragment (32 B)
union H8   { unsigned short h[8]; v4u q; };   // 8 packed bf16 (16 B)

__device__ __forceinline__ unsigned short f2bf(float f) {
    unsigned u = __builtin_bit_cast(unsigned, f);
    u += 0x7FFFu + ((u >> 16) & 1u);          // round-to-nearest-even
    return (unsigned short)(u >> 16);
}

// K-offset (within a 32-wide chunk) for fragment slot j, given lane half.
// ISA 7.12.2, 16-bit A 16x32: lane<16 -> K {0..7,16..23}; lane>=16 -> K {8..15,24..31}.
__device__ __forceinline__ int frag_k(int j, int half) {
    int base = half ? 8 : 0;
    return (j < 8) ? (base + j) : (base + 8 + j);
}

// ---------------------------------------------------------------------------
// Pack [W1 | W2] (row-major [Nout][K], fp32) into bf16 WMMA B-fragments.
// Virtual K axis: [0,Kpad1) -> W1 (zero pad past K1), [Kpad1,Kpad1+Kpad2) -> W2.
// Output layout: frag t = ((nt*KC + kc)*32 + lane), 16 bf16 each.
// ---------------------------------------------------------------------------
__global__ void pack_frags(const float* __restrict__ W1, const float* __restrict__ W2,
                           unsigned short* __restrict__ out,
                           int Nout, int K1, int Kpad1, int K2, int Kpad2) {
    int KC = (Kpad1 + Kpad2) >> 5;
    int t = blockIdx.x * blockDim.x + threadIdx.x;
    int total = (Nout >> 4) * KC * 32;
    if (t >= total) return;
    int lane = t & 31;
    int kc   = (t >> 5) % KC;
    int nt   = (t >> 5) / KC;
    int half = lane >> 4;
    int n    = nt * 16 + (lane & 15);

    H8 lo, hi;
#pragma unroll
    for (int j = 0; j < 16; ++j) {
        int k = kc * 32 + frag_k(j, half);
        float v = 0.f;
        if (k < Kpad1) { if (k < K1) v = W1[(long long)n * K1 + k]; }
        else { int k2 = k - Kpad1; if (k2 < K2) v = W2[(long long)n * K2 + k2]; }
        unsigned short bf = f2bf(v);
        if (j < 8) lo.h[j] = bf; else hi.h[j - 8] = bf;
    }
    v4u* dst = (v4u*)(out + (size_t)t * 16);
    dst[0] = lo.q;
    dst[1] = hi.q;
}

// ---------------------------------------------------------------------------
// agg[dst] += x[src] * ew   (segment_sum of messages)
// ---------------------------------------------------------------------------
__global__ void scatter_edges(const float* __restrict__ x,
                              const long long* __restrict__ ei,
                              const float* __restrict__ ew,
                              float* __restrict__ agg, int D) {
    long long t = (long long)blockIdx.x * blockDim.x + threadIdx.x;
    long long tot = (long long)NE * D;
    if (t >= tot) return;
    int e = (int)(t / D);
    int d = (int)(t % D);
    long long s  = ei[e];
    long long dn = ei[NE + e];
    float v = x[s * D + d] * ew[e];
    atomicAdd(&agg[dn * D + d], v);
}

// ---------------------------------------------------------------------------
// out = act( agg @ Wrel^T + brel + x @ Wroot^T )
// One wave per 16-node tile; virtual K = [agg | x] so one WMMA K-loop covers
// both GEMMs. A staged in LDS in fragment order (ds_load_b128 reads);
// B fragments pre-packed in global (global_load_b128 reads).
// ---------------------------------------------------------------------------
__global__ void __launch_bounds__(64)
node_conv_wmma(const float* __restrict__ agg, const float* __restrict__ x,
               const unsigned short* __restrict__ pW,
               const float* __restrict__ brel, float* __restrict__ out,
               int Din, int Dout, int act) {
    __shared__ __attribute__((aligned(16))) unsigned short sA[2][16][32][16]; // 32 KB
    const int lane = threadIdx.x & 31;
    const int wv   = threadIdx.x >> 5;
    const int half = lane >> 4;
    const int l16  = lane & 15;
    const int mbase = (blockIdx.x * 2 + wv) * 16;
    const int Kpad = (Din + 31) & ~31;
    const int KC   = (2 * Kpad) >> 5;         // chunks over [agg | x]

    // Stage A in fragment order: sA[wv][kc][lane][j] = bf16 of (row=lane&15,
    // k=kc*32+frag_k(j,lane>>4)). Each thread-iteration writes 8 bf16 (16 B).
    for (int g = lane; g < KC * 64; g += 32) {
        int kc = g >> 6, r = g & 63, tl = r >> 1, jg = r & 1;
        int row = tl & 15, th = tl >> 4;
        int node = mbase + row;
        int kbase = kc * 32 + (th ? 8 : 0) + (jg ? 16 : 0);
        H8 hh;
#pragma unroll
        for (int j8 = 0; j8 < 8; ++j8) {
            int k = kbase + j8;
            float v = 0.f;
            if (k < Kpad) { if (k < Din) v = agg[(long long)node * Din + k]; }
            else { int k2 = k - Kpad; if (k2 < Din) v = x[(long long)node * Din + k2]; }
            hh.h[j8] = f2bf(v);
        }
        *(v4u*)&sA[wv][kc][tl][jg * 8] = hh.q;
    }
    __syncthreads();

    for (int nt = 0; nt < Dout / 16; ++nt) {
        int n = nt * 16 + l16;
        v8f c = {};
        for (int kc = 0; kc < KC; ++kc) {
            Frag a, b;
            const v4u* ap = (const v4u*)&sA[wv][kc][lane][0];
            a.q[0] = ap[0]; a.q[1] = ap[1];
            const v4u* bp = (const v4u*)(pW + (((size_t)nt * KC + kc) * 32 + lane) * 16);
            b.q[0] = bp[0]; b.q[1] = bp[1];
            c = __builtin_amdgcn_wmma_f32_16x16x32_bf16(false, a.v, false, b.v,
                                                        (short)0, c, false, false);
        }
        if (nt + 1 < Dout / 16)
            __builtin_prefetch(pW + (((size_t)(nt + 1) * KC) * 32 + lane) * 16, 0, 0);
        float bn = brel[n];
#pragma unroll
        for (int i = 0; i < 8; ++i) {
            int m = half ? (8 + i) : i;            // C/D layout: lanes 16-31 hold M=8..15
            float v = c[i] + bn;
            if (act) v = v > 0.f ? v : 0.f;
            out[(long long)(mbase + m) * Dout + n] = v;
        }
    }
}

// ---------------------------------------------------------------------------
// new_ea = relu(feat @ fW^T + fb) @ rW^T + rb   (second layer fused)
// feat = [ (x_i-x_j)^2 for d<D-1 | x_i[D-1] | ea ]; one wave per 16 edges.
// ---------------------------------------------------------------------------
__global__ void __launch_bounds__(128)
edge_mlp_wmma(const float* __restrict__ x, const long long* __restrict__ ei,
              const float* __restrict__ ea,
              const unsigned short* __restrict__ pW,
              const float* __restrict__ fb, const float* __restrict__ rW,
              const float* __restrict__ rb, float* __restrict__ out,
              int D, int H) {
    __shared__ __attribute__((aligned(16))) unsigned short sF[4][9][32][16]; // 36 KB
    const int lane = threadIdx.x & 31;
    const int wv   = threadIdx.x >> 5;
    const int half = lane >> 4;
    const int l16  = lane & 15;
    const int ebase = (blockIdx.x * 4 + wv) * 16;
    const int K    = D + 1;
    const int Kpad = (K + 31) & ~31;
    const int KC   = Kpad >> 5;

    // Stage feature tile in fragment order (8 bf16 per ds_store_b128).
    for (int g = lane; g < KC * 64; g += 32) {
        int kc = g >> 6, r = g & 63, tl = r >> 1, jg = r & 1;
        int row = tl & 15, th = tl >> 4;
        int e = ebase + row;
        long long s = ei[e], d2 = ei[NE + e];
        int kbase = kc * 32 + (th ? 8 : 0) + (jg ? 16 : 0);
        H8 hh;
#pragma unroll
        for (int j8 = 0; j8 < 8; ++j8) {
            int k = kbase + j8;
            float v;
            if (k < D - 1)       { float df = x[s * D + k] - x[d2 * D + k]; v = df * df; }
            else if (k == D - 1)   v = x[s * D + (D - 1)];
            else if (k == D)       v = ea[e];
            else                   v = 0.f;
            hh.h[j8] = f2bf(v);
        }
        *(v4u*)&sF[wv][kc][tl][jg * 8] = hh.q;
    }
    __syncthreads();

    float part[8];
#pragma unroll
    for (int i = 0; i < 8; ++i) part[i] = 0.f;

    for (int nt = 0; nt < H / 16; ++nt) {
        int n = nt * 16 + l16;
        v8f c = {};
        for (int kc = 0; kc < KC; ++kc) {
            Frag a, b;
            const v4u* ap = (const v4u*)&sF[wv][kc][lane][0];
            a.q[0] = ap[0]; a.q[1] = ap[1];
            const v4u* bp = (const v4u*)(pW + (((size_t)nt * KC + kc) * 32 + lane) * 16);
            b.q[0] = bp[0]; b.q[1] = bp[1];
            c = __builtin_amdgcn_wmma_f32_16x16x32_bf16(false, a.v, false, b.v,
                                                        (short)0, c, false, false);
        }
        if (nt + 1 < H / 16)
            __builtin_prefetch(pW + (((size_t)(nt + 1) * KC) * 32 + lane) * 16, 0, 0);
        float fbn = fb[n], rwn = rW[n];
#pragma unroll
        for (int i = 0; i < 8; ++i) {
            float h = c[i] + fbn;
            h = h > 0.f ? h : 0.f;
            part[i] += h * rwn;                    // fused [H,1] second layer
        }
    }
    // Reduce over the 16 N-lanes within each half (halves hold M=0..7 / 8..15)
#pragma unroll
    for (int mask = 1; mask <= 8; mask <<= 1)
#pragma unroll
        for (int i = 0; i < 8; ++i) part[i] += __shfl_xor(part[i], mask, 32);

    float rbv = rb[0];
    if (l16 == 0) {
        int moff = half ? 8 : 0;
#pragma unroll
        for (int i = 0; i < 8; ++i) out[ebase + moff + i] = part[i] + rbv;
    }
}

// ---------------------------------------------------------------------------
extern "C" void kernel_launch(void* const* d_in, const int* in_sizes, int n_in,
                              void* d_out, int out_size, void* d_ws, size_t ws_size,
                              hipStream_t stream) {
    (void)in_sizes; (void)n_in; (void)out_size; (void)ws_size;

    const float*     x0  = (const float*)d_in[0];
    const long long* ei  = (const long long*)d_in[1];
    const float*     ea0 = (const float*)d_in[2];

    const float* Wrel[4]  = {(const float*)d_in[3],  (const float*)d_in[6],
                             (const float*)d_in[9],  (const float*)d_in[12]};
    const float* brel[4]  = {(const float*)d_in[4],  (const float*)d_in[7],
                             (const float*)d_in[10], (const float*)d_in[13]};
    const float* Wroot[4] = {(const float*)d_in[5],  (const float*)d_in[8],
                             (const float*)d_in[11], (const float*)d_in[14]};
    const float* fW[4]    = {(const float*)d_in[15], (const float*)d_in[19],
                             (const float*)d_in[23], (const float*)d_in[27]};
    const float* fb[4]    = {(const float*)d_in[16], (const float*)d_in[20],
                             (const float*)d_in[24], (const float*)d_in[28]};
    const float* rW[4]    = {(const float*)d_in[17], (const float*)d_in[21],
                             (const float*)d_in[25], (const float*)d_in[29]};
    const float* rb[4]    = {(const float*)d_in[18], (const float*)d_in[22],
                             (const float*)d_in[26], (const float*)d_in[30]};

    const int Din[4]  = {24, 128, 256, 64};
    const int Dout[4] = {128, 256, 64, 16};
    const int Hd[4]   = {64, 128, 64, 32};

    char*  ws  = (char*)d_ws;
    float* xA  = (float*)ws;                                     // 100k x 256 f32
    float* xB  = (float*)(ws + (size_t)NN * 256 * 4);
    float* agg = (float*)(ws + (size_t)NN * 256 * 8);
    float* eaA = (float*)(ws + (size_t)NN * 256 * 12);
    float* eaB = (float*)(ws + (size_t)NN * 256 * 12 + (size_t)NE * 4);
    unsigned short* pWn = (unsigned short*)(ws + (size_t)NN * 256 * 12 + (size_t)NE * 8);
    unsigned short* pWe = pWn + 256 * 512;                       // node pack <= 256 KB

    float* outx  = (float*)d_out;                                // [100k,16]
    float* outea = (float*)d_out + (size_t)NN * 16;              // [800k]

    const float* xcur[4]  = {x0, xA, xB, xA};
    float*       xnext[4] = {xA, xB, xA, outx};
    const float* eacur[4] = {ea0, eaA, eaB, eaA};
    float*       eanxt[4] = {eaA, eaB, eaA, outea};

    for (int L = 0; L < 4; ++L) {
        int D = Din[L], Do = Dout[L], H = Hd[L];
        int KpadN = (D + 31) & ~31;
        int Ke = D + 1, KpadE = (Ke + 31) & ~31;

        // Pack weights into WMMA fragment order (bf16), once per layer.
        int totN = (Do / 16) * ((2 * KpadN) / 32) * 32;
        pack_frags<<<(totN + 255) / 256, 256, 0, stream>>>(
            Wrel[L], Wroot[L], pWn, Do, D, KpadN, D, KpadN);
        int totE = (H / 16) * (KpadE / 32) * 32;
        pack_frags<<<(totE + 255) / 256, 256, 0, stream>>>(
            fW[L], fW[L], pWe, H, Ke, KpadE, 0, 0);

        hipMemsetAsync(agg, 0, (size_t)NN * D * sizeof(float), stream);

        long long tot = (long long)NE * D;
        int sblocks = (int)((tot + 255) / 256);
        scatter_edges<<<sblocks, 256, 0, stream>>>(xcur[L], ei, eacur[L], agg, D);

        edge_mlp_wmma<<<NE / 16 / 4, 128, 0, stream>>>(
            xcur[L], ei, eacur[L], pWe, fb[L], rW[L], rb[L], eanxt[L], D, H);

        node_conv_wmma<<<NN / 16 / 2, 64, 0, stream>>>(
            agg, xcur[L], pWn, brel[L], xnext[L], D, Do, (L < 3) ? 1 : 0);
    }
}
